// MoE_9947144258207
// MI455X (gfx1250) — matrix operand order repaired and checked by reading, not compiled, not run on hardware
//
#include <hip/hip_runtime.h>
#include <hip/hip_bf16.h>

// Problem constants (B=4, S=2048, D=1024, F=4096, E=8, TOP_K=2)
#define N_TOK   8192
#define D_MODEL 1024
#define D_FF    4096
#define N_EXP   8

typedef __attribute__((ext_vector_type(16))) __bf16 v16bf;
typedef __attribute__((ext_vector_type(8)))  __bf16 v8bf;
typedef __attribute__((ext_vector_type(8)))  float  v8f;
typedef __attribute__((ext_vector_type(4)))  float  v4f;
typedef __attribute__((ext_vector_type(4)))  unsigned int u32x4;
typedef __attribute__((ext_vector_type(8)))  int i32x8;
typedef __attribute__((ext_vector_type(4)))  int i32x4;

#if __has_builtin(__builtin_amdgcn_tensor_load_to_lds)
#define USE_TDM 1
#else
#define USE_TDM 0
#endif

// ---------------------------------------------------------------------------
// 0) zero output + expert counters
// ---------------------------------------------------------------------------
__global__ void zero_init_kernel(float* __restrict__ out, int* __restrict__ counts,
                                 size_t n_out) {
  size_t i = (size_t)blockIdx.x * blockDim.x + threadIdx.x;
  size_t stride = (size_t)gridDim.x * blockDim.x;
  for (size_t j = i; j < n_out; j += stride) out[j] = 0.0f;
  if (i < N_EXP) counts[i] = 0;
}

// ---------------------------------------------------------------------------
// 1) fp32 -> bf16 conversion (vectorized: 8 elems / thread / iter)
// ---------------------------------------------------------------------------
__global__ void cvt_f32_bf16_kernel(const float* __restrict__ src,
                                    __bf16* __restrict__ dst, size_t n8) {
  size_t i = (size_t)blockIdx.x * blockDim.x + threadIdx.x;
  size_t stride = (size_t)gridDim.x * blockDim.x;
  for (size_t j = i; j < n8; j += stride) {
    const v4f* s = (const v4f*)(src + j * 8);
    v4f a = s[0];
    v4f b = s[1];
    v8bf o;
    o[0] = (__bf16)a[0]; o[1] = (__bf16)a[1]; o[2] = (__bf16)a[2]; o[3] = (__bf16)a[3];
    o[4] = (__bf16)b[0]; o[5] = (__bf16)b[1]; o[6] = (__bf16)b[2]; o[7] = (__bf16)b[3];
    *(v8bf*)(dst + j * 8) = o;
  }
}

// ---------------------------------------------------------------------------
// 2) Router: one wave32 per token. 8 dot products, softmax, top-2,
//    renormalized gates, atomic append into per-expert token lists.
// ---------------------------------------------------------------------------
__global__ void router_kernel(const float* __restrict__ x,
                              const float* __restrict__ rw,
                              const float* __restrict__ rb,
                              int*   __restrict__ counts,
                              int*   __restrict__ tok_idx,
                              float* __restrict__ tok_gate) {
  const int wave = threadIdx.x >> 5;
  const int lane = threadIdx.x & 31;
  const int t = blockIdx.x * 8 + wave;
  if (t >= N_TOK) return;

  float logits[N_EXP];
#pragma unroll
  for (int e = 0; e < N_EXP; ++e) {
    float acc = 0.0f;
    const float* xp = x + (size_t)t * D_MODEL + lane;
    const float* wp = rw + (size_t)e * D_MODEL + lane;
#pragma unroll 8
    for (int j = 0; j < D_MODEL / 32; ++j) acc += xp[j * 32] * wp[j * 32];
    for (int off = 16; off > 0; off >>= 1) acc += __shfl_xor(acc, off, 32);
    logits[e] = acc + rb[e];
  }

  float m = logits[0];
#pragma unroll
  for (int e = 1; e < N_EXP; ++e) m = fmaxf(m, logits[e]);
  float p[N_EXP];
  float s = 0.0f;
#pragma unroll
  for (int e = 0; e < N_EXP; ++e) { p[e] = __expf(logits[e] - m); s += p[e]; }
  float inv_s = 1.0f / s;
#pragma unroll
  for (int e = 0; e < N_EXP; ++e) p[e] *= inv_s;

  float best1 = -1.0f, best2 = -1.0f;
  int i1 = 0, i2 = 0;
#pragma unroll
  for (int e = 0; e < N_EXP; ++e) {
    bool gt1 = p[e] > best1;
    bool gt2 = p[e] > best2;
    float nb2 = gt1 ? best1 : (gt2 ? p[e] : best2);
    int   ni2 = gt1 ? i1    : (gt2 ? e    : i2);
    best1 = gt1 ? p[e] : best1;
    i1    = gt1 ? e    : i1;
    best2 = nb2; i2 = ni2;
  }
  float denom = best1 + best2;
  float g1 = best1 / denom;
  float g2 = best2 / denom;

  if (lane == 0) {
    int pos = atomicAdd(&counts[i1], 1);
    tok_idx[i1 * N_TOK + pos] = t;
    tok_gate[i1 * N_TOK + pos] = g1;
    pos = atomicAdd(&counts[i2], 1);
    tok_idx[i2 * N_TOK + pos] = t;
    tok_gate[i2 * N_TOK + pos] = g2;
  }
}

// ---------------------------------------------------------------------------
// 3) Fused SwiGLU FFN per (expert, 32-token tile), bf16 WMMA.
//    x-tile gathered into LDS by the Tensor Data Mover (gather-mode D#),
//    TDM pad: 4 dwords per 1KB -> row stride 2080B, +16B shift mid-row.
// ---------------------------------------------------------------------------
#define TM     32      // tokens per tile (2 WMMA M-tiles)
#define FC     128     // f-chunk width (K for second GEMM)
#define HSTR   136     // padded bf16 row stride for LDS h buffer (68 dwords)
#define XROWB  2080    // bytes per LDS x row: 2048 + 2 pads of 16B

#define XS_BYTES   (TM * XROWB)             // 66560
#define HB_BYTES   (TM * HSTR * 2)          // 8704
#define SMEM_BYTES (XS_BYTES + HB_BYTES + 256)

__device__ __forceinline__ v8f wmma_bf16(v16bf a, v16bf b, v8f c) {
  return __builtin_amdgcn_wmma_f32_16x16x32_bf16(
      /*neg_a=*/false, a, /*neg_b=*/false, b,
      /*c_mod=*/(short)0, c, /*reuse_a=*/false, /*reuse_b=*/false);
}

// A-matrix fragment: contiguous k [k0..k0+15] per lane (this code's order, B matches)
__device__ __forceinline__ v16bf load_a16(const __bf16* p) {
  v8bf lo = *(const v8bf*)(p);
  v8bf hi = *(const v8bf*)(p + 8);
  v16bf r;
#pragma unroll
  for (int i = 0; i < 8; ++i) { r[i] = lo[i]; r[i + 8] = hi[i]; }
  return r;
}

#if USE_TDM
#define RFL(v) __builtin_amdgcn_readfirstlane(v)
// Issue one gather-mode TENSOR_LOAD_TO_LDS: 16 rows of 2KB from xb into LDS.
__device__ __forceinline__ void tdm_gather16(const __bf16* xb, unsigned lds_addr,
                                             const int* s_tok, int base) {
  u32x4 g0;
  i32x8 g1;
  i32x4 g2, g3;
  unsigned long long ga = (unsigned long long)(uintptr_t)xb;
  g0[0] = 0x80000001u;                 // count=1, gather_mode=1, 16-bit indices
  g0[1] = lds_addr;                    // LDS byte address of tile start
  g0[2] = (unsigned)ga;                // global_addr[31:0]
  g0[3] = (unsigned)((ga >> 32) & 0x1FFFFFFu) | 0x80000000u;  // addr[56:32], type=2
  // data_size=3 (8B units), pad_enable, pad_interval=7 (256 dw), pad_amount=3 (4 dw)
  g1[0] = (3 << 16) | (1 << 20) | (7 << 22) | (3 << 25);
  g1[1] = (256u & 0xFFFFu) << 16;      // tensor_dim0[15:0]   (row = 256 units)
  g1[2] = (unsigned)(N_TOK & 0xFFFF) << 16;   // tensor_dim1[15:0] = 8192 rows
  g1[3] = 256u << 16;                  // tile_dim0 = 256 units (2KB row)
  g1[4] = 16;                          // tile_dim1 = #valid gather indices
  g1[5] = 256;                         // tensor_dim0_stride = 256 units
  g1[6] = 0;
  g1[7] = 0;
  int t0  = RFL(s_tok[base + 0]),  t1  = RFL(s_tok[base + 1]);
  int t2  = RFL(s_tok[base + 2]),  t3  = RFL(s_tok[base + 3]);
  int t4  = RFL(s_tok[base + 4]),  t5  = RFL(s_tok[base + 5]);
  int t6  = RFL(s_tok[base + 6]),  t7  = RFL(s_tok[base + 7]);
  int t8  = RFL(s_tok[base + 8]),  t9  = RFL(s_tok[base + 9]);
  int t10 = RFL(s_tok[base + 10]), t11 = RFL(s_tok[base + 11]);
  int t12 = RFL(s_tok[base + 12]), t13 = RFL(s_tok[base + 13]);
  int t14 = RFL(s_tok[base + 14]), t15 = RFL(s_tok[base + 15]);
  g2[0] = (t1 << 16)  | (t0 & 0xFFFF);
  g2[1] = (t3 << 16)  | (t2 & 0xFFFF);
  g2[2] = (t5 << 16)  | (t4 & 0xFFFF);
  g2[3] = (t7 << 16)  | (t6 & 0xFFFF);
  g3[0] = (t9 << 16)  | (t8 & 0xFFFF);
  g3[1] = (t11 << 16) | (t10 & 0xFFFF);
  g3[2] = (t13 << 16) | (t12 & 0xFFFF);
  g3[3] = (t15 << 16) | (t14 & 0xFFFF);
#if defined(__clang_major__) && (__clang_major__ >= 23)
  i32x8 g4 = (i32x8)(0);
  __builtin_amdgcn_tensor_load_to_lds(g0, g1, g2, g3, g4, 0);
#else
  __builtin_amdgcn_tensor_load_to_lds(g0, g1, g2, g3, 0);
#endif
}
#endif  // USE_TDM

__global__ __launch_bounds__(256)
void moe_ffn_kernel(const __bf16* __restrict__ xb,
                    const __bf16* __restrict__ w1b,
                    const __bf16* __restrict__ w3b,
                    const __bf16* __restrict__ w2b,
                    const int*    __restrict__ counts,
                    const int*    __restrict__ tok_idx,
                    const float*  __restrict__ tok_gate,
                    float*        __restrict__ out) {
  extern __shared__ char smem[];
  char*   xs     = smem;                                   // [TM][XROWB bytes]
  __bf16* hbuf   = (__bf16*)(smem + XS_BYTES);             // [TM][HSTR]
  int*    s_tok  = (int*)  (smem + XS_BYTES + HB_BYTES);   // [TM]
  float*  s_gate = (float*)(smem + XS_BYTES + HB_BYTES + 128);

  const int e = blockIdx.y;
  const int tile = blockIdx.x;
  const int cnt = counts[e];
  if (tile * TM >= cnt) return;   // empty tile: cheap early exit

  if (threadIdx.x < TM) {
    int slot = tile * TM + threadIdx.x;
    if (slot < cnt) {
      s_tok[threadIdx.x]  = tok_idx[e * N_TOK + slot];
      s_gate[threadIdx.x] = tok_gate[e * N_TOK + slot];
    } else {
      s_tok[threadIdx.x]  = 0;     // safe dummy row
      s_gate[threadIdx.x] = 0.0f;  // contributes nothing
    }
  }
  __syncthreads();

  // ---- stage gathered x rows into LDS (TDM gather or manual fallback) ----
#if USE_TDM
  if (threadIdx.x < 32) {          // wave 0 issues both DMA descriptors
    unsigned lds0 = (unsigned)(uintptr_t)(void*)xs;
    tdm_gather16(xb, lds0, s_tok, 0);
    tdm_gather16(xb, lds0 + 16u * XROWB, s_tok, 16);
    __builtin_amdgcn_s_wait_tensorcnt(0);
  }
  __syncthreads();
#else
  {
    const int tid = threadIdx.x;
#pragma unroll
    for (int it = 0; it < (TM * (D_MODEL / 8)) / 256; ++it) {  // 16 iters
      int c = it * 256 + tid;
      int row = c >> 7;            // 128 x 16B chunks per row
      int col8 = c & 127;
      v8bf v = *(const v8bf*)(xb + (size_t)s_tok[row] * D_MODEL + col8 * 8);
      *(v8bf*)(xs + row * XROWB + col8 * 16 + (col8 >= 64 ? 16 : 0)) = v;
    }
  }
  __syncthreads();
#endif

  const int wave  = threadIdx.x >> 5;
  const int lane  = threadIdx.x & 31;
  const int lhalf = lane & 15;
  const bool hi   = lane >= 16;
  const int aoff  = hi ? 16 : 0;   // A-fragment K offset per lane half
  const int boff  = hi ? 16 : 0;   // B-fragment K offset per lane half

  const __bf16* w1e = w1b + (size_t)e * D_FF * D_MODEL;
  const __bf16* w3e = w3b + (size_t)e * D_FF * D_MODEL;
  const __bf16* w2e = w2b + (size_t)e * D_MODEL * D_FF;

  v8f acc[2][8];
#pragma unroll
  for (int m = 0; m < 2; ++m)
#pragma unroll
    for (int n = 0; n < 8; ++n) acc[m][n] = (v8f)(0.0f);

  for (int fc = 0; fc < D_FF / FC; ++fc) {
    const int fcol = fc * FC + wave * 16;

    // ---- prefetch next chunk's streaming B panels while WMMAs run ----
    {
      const int fcn = (fc + 1) & (D_FF / FC - 1);
      const __bf16* p1 = w1e + (size_t)(fcn * FC + wave * 16) * D_MODEL;
      const __bf16* p3 = w3e + (size_t)(fcn * FC + wave * 16) * D_MODEL;
#pragma unroll
      for (int i = 0; i < 8; ++i) {   // 16 rows x 2KB contiguous = 32KB each
        __builtin_prefetch(p1 + (size_t)i * 2048 + lane * 64, 0, 3);
        __builtin_prefetch(p3 + (size_t)i * 2048 + lane * 64, 0, 3);
      }
#pragma unroll
      for (int i = 0; i < 4; ++i) {   // 128 strided w2 rows, 256B each
        __builtin_prefetch(w2e + (size_t)(wave * 128 + i * 32 + lane) * D_FF
                               + fcn * FC, 0, 3);
      }
    }

    // ---- phase 1: h strip cols [fcol, fcol+16), A from LDS, B from global ----
    v8f a1_0 = (v8f)(0.0f), a1_1 = (v8f)(0.0f);
    v8f a3_0 = (v8f)(0.0f), a3_1 = (v8f)(0.0f);
    const __bf16* w1p = w1e + (size_t)(fcol + lhalf) * D_MODEL + boff;
    const __bf16* w3p = w3e + (size_t)(fcol + lhalf) * D_MODEL + boff;
    const char* xl0 = xs + lhalf * XROWB + aoff * 2;
    const char* xl1 = xs + (16 + lhalf) * XROWB + aoff * 2;
#pragma unroll
    for (int kh = 0; kh < 2; ++kh) {       // TDM pad: +16B after first 1KB
      const int xsh = kh ? 16 : 0;
#pragma unroll 4
      for (int k2 = 0; k2 < 16; ++k2) {
        const int ks = kh * 16 + k2;
        v16bf b1 = *(const v16bf*)(w1p + ks * 32);
        v16bf b3 = *(const v16bf*)(w3p + ks * 32);
        v16bf a0 = load_a16((const __bf16*)(xl0 + ks * 64 + xsh));
        v16bf a1 = load_a16((const __bf16*)(xl1 + ks * 64 + xsh));
        a1_0 = wmma_bf16(a0, b1, a1_0);
        a3_0 = wmma_bf16(a0, b3, a3_0);
        a1_1 = wmma_bf16(a1, b1, a1_1);
        a3_1 = wmma_bf16(a1, b3, a3_1);
      }
    }
    // silu(a1)*a3 -> bf16 into LDS (C layout: M = i + (hi?8:0), N = lhalf)
    {
      const int rb = hi ? 8 : 0;
      const int hcol = wave * 16 + lhalf;
#pragma unroll
      for (int i = 0; i < 8; ++i) {
        float s0 = a1_0[i];
        float h0 = (s0 / (1.0f + __expf(-s0))) * a3_0[i];
        float s1 = a1_1[i];
        float h1 = (s1 / (1.0f + __expf(-s1))) * a3_1[i];
        hbuf[(rb + i) * HSTR + hcol]      = (__bf16)h0;
        hbuf[(16 + rb + i) * HSTR + hcol] = (__bf16)h1;
      }
    }
    __syncthreads();

    // ---- phase 2: y[:, wave*128 .. +128) += h(32xFC) @ W2^T ----
    const __bf16* w2p = w2e + fc * FC + boff;
#pragma unroll
    for (int ks = 0; ks < FC / 32; ++ks) {
      v16bf ah0 = load_a16(&hbuf[lhalf * HSTR + ks * 32 + aoff]);
      v16bf ah1 = load_a16(&hbuf[(16 + lhalf) * HSTR + ks * 32 + aoff]);
#pragma unroll
      for (int n = 0; n < 8; ++n) {
        const int ncol = wave * 128 + n * 16 + lhalf;
        v16bf b2 = *(const v16bf*)(w2p + (size_t)ncol * D_FF + ks * 32);
        acc[0][n] = wmma_bf16(ah0, b2, acc[0][n]);
        acc[1][n] = wmma_bf16(ah1, b2, acc[1][n]);
      }
    }
    __syncthreads();  // hbuf reused next chunk
  }

  // ---- epilogue: gate-scale + scatter-add into out ----
  const int rb = hi ? 8 : 0;
#pragma unroll
  for (int m = 0; m < 2; ++m) {
#pragma unroll
    for (int i = 0; i < 8; ++i) {
      const int row = m * 16 + rb + i;
      const float g = s_gate[row];
      float* op = out + (size_t)s_tok[row] * D_MODEL + wave * 128 + lhalf;
#pragma unroll
      for (int n = 0; n < 8; ++n) atomicAdd(op + n * 16, g * acc[m][n][i]);
    }
  }
}

// ---------------------------------------------------------------------------
// Launch
// ---------------------------------------------------------------------------
extern "C" void kernel_launch(void* const* d_in, const int* in_sizes, int n_in,
                              void* d_out, int out_size, void* d_ws, size_t ws_size,
                              hipStream_t stream) {
  (void)in_sizes; (void)n_in; (void)out_size; (void)ws_size;
  const float* x  = (const float*)d_in[0];
  const float* rw = (const float*)d_in[1];
  const float* rb = (const float*)d_in[2];
  const float* w1 = (const float*)d_in[3];
  const float* w2 = (const float*)d_in[4];
  const float* w3 = (const float*)d_in[5];
  float* out = (float*)d_out;

  // workspace layout (256B aligned)
  char* ws = (char*)d_ws;
  size_t off = 0;
  auto carve = [&](size_t bytes) -> void* {
    void* p = ws + off;
    off = (off + bytes + 255) & ~(size_t)255;
    return p;
  };
  const size_t n_x  = (size_t)N_TOK * D_MODEL;
  const size_t n_w  = (size_t)N_EXP * D_FF * D_MODEL;
  __bf16* xb  = (__bf16*)carve(n_x * 2);
  __bf16* w1b = (__bf16*)carve(n_w * 2);
  __bf16* w3b = (__bf16*)carve(n_w * 2);
  __bf16* w2b = (__bf16*)carve(n_w * 2);
  int*    counts   = (int*)carve(N_EXP * 4);
  int*    tok_idx  = (int*)carve((size_t)N_EXP * N_TOK * 4);
  float*  tok_gate = (float*)carve((size_t)N_EXP * N_TOK * 4);

  // 0) zero out + counters
  zero_init_kernel<<<2048, 256, 0, stream>>>(out, counts, n_x);

  // 1) fp32 -> bf16
  cvt_f32_bf16_kernel<<<(unsigned)(n_x / 8 / 256), 256, 0, stream>>>(x,  xb,  n_x / 8);
  cvt_f32_bf16_kernel<<<16384, 256, 0, stream>>>(w1, w1b, n_w / 8);
  cvt_f32_bf16_kernel<<<16384, 256, 0, stream>>>(w3, w3b, n_w / 8);
  cvt_f32_bf16_kernel<<<16384, 256, 0, stream>>>(w2, w2b, n_w / 8);

  // 2) router + gather
  router_kernel<<<N_TOK / 8, 256, 0, stream>>>(x, rw, rb, counts, tok_idx, tok_gate);

  // 3) expert FFN (worst-case grid; empty tiles exit immediately)
  dim3 grid(N_TOK / TM, N_EXP);
  moe_ffn_kernel<<<grid, 256, SMEM_BYTES, stream>>>(xb, w1b, w3b, w2b,
                                                    counts, tok_idx, tok_gate, out);
}
